// GatedDeltaNet_40218073760165
// MI455X (gfx1250) — compile-verified, matrix operand order checked by reading, not run on hardware
//
#include <hip/hip_runtime.h>
#include <hip/hip_bf16.h>
#include <math.h>

// ---------------- problem constants (from reference) ----------------
#define BB    2
#define TT    2048
#define HID   2048
#define NKH   16
#define NVH   32
#define HD    128
#define KW    4
#define KDIM  (NKH*HD)        // 2048
#define VDIM  (NVH*HD)        // 4096
#define CONVC (2*KDIM+VDIM)   // 8192
#define EXP_  (NVH/NKH)       // 2
#define MROWS (BB*TT)         // 4096

// ---------------- WMMA types ----------------
typedef __bf16 bf16;
typedef __attribute__((ext_vector_type(16))) __bf16        bf16x16;
typedef __attribute__((ext_vector_type(8)))  float         f32x8;
typedef __attribute__((ext_vector_type(8)))  unsigned int  uint8v;

union FragU { uint8v u; bf16x16 v; };
union Pack4 { bf16 h[4]; uint2 u2; };

// gfx1250 async global->LDS copy (16B per lane), tracked by ASYNCcnt.
__device__ __forceinline__ void async_copy16(unsigned int lds_off,
                                             const void* gaddr) {
  asm volatile("global_load_async_to_lds_b128 %0, %1, off"
               :: "v"(lds_off), "v"((unsigned long long)(uintptr_t)gaddr)
               : "memory");
}
__device__ __forceinline__ void wait_async0() {
  asm volatile("s_wait_asynccnt 0x0" ::: "memory");
}

// =====================================================================
// fp32 -> bf16 pack (4 elements / thread)
// =====================================================================
__global__ void cvt_bf16_kernel(const float* __restrict__ in,
                                bf16* __restrict__ outp, size_t n) {
  size_t i = ((size_t)blockIdx.x * blockDim.x + threadIdx.x) * 4;
  if (i + 3 < n) {
    float4 v = *(const float4*)&in[i];
    Pack4 p;
    p.h[0] = (bf16)v.x; p.h[1] = (bf16)v.y;
    p.h[2] = (bf16)v.z; p.h[3] = (bf16)v.w;
    *(uint2*)&outp[i] = p.u2;
  } else {
    for (; i < n; ++i) outp[i] = (bf16)in[i];
  }
}

// =====================================================================
// bf16 WMMA GEMM:  Y[M,N] = X[M,K] * W[N,K]^T   (X,W bf16; Y fp32)
// Block tile 128(M) x 64(N), K-step 64, 256 threads = 8 waves.
// Wave (w&3) -> 32-row M sub-tile, (w>>2) -> 32-col N sub-tile,
// 2x2 accumulators -> 8 WMMAs per K-step.
// Tiles move global->LDS via gfx1250 async-LDS DMA, double-buffered:
// DMA of tile k+1 overlaps WMMA on tile k; one s_wait_asynccnt +
// one barrier per K-step. Out-of-range B rows are clamp-loaded
// (their output columns are never stored).
// M % 128 == 0 (4096), K % 64 == 0; N guarded at store.
// =====================================================================
__global__ __launch_bounds__(256)
void gemm_bf16_wmma(const bf16* __restrict__ X, const bf16* __restrict__ W,
                    float* __restrict__ Y, int M, int N, int Kd) {
  (void)M;
  __shared__ __align__(32) bf16 As[2][128][64];   // 2 x 16 KB
  __shared__ __align__(32) bf16 Bs[2][64][64];    // 2 x  8 KB

  const int tid     = threadIdx.x;
  const int wave    = tid >> 5;
  const int lane    = tid & 31;
  const int lane_lo = lane & 15;
  const int lane_hi = lane >> 4;

  const int mBlock = blockIdx.y * 128;
  const int nBlock = blockIdx.x * 64;
  const int mWave  = (wave & 3) * 32;
  const int nWave  = (wave >> 2) * 32;

  // 16B chunks: a row of 64 bf16 = 8 chunks
  const int crow = tid >> 3;            // 0..31
  const int ccol = (tid & 7) * 8;       // element col of this chunk

  const unsigned int ldsA = (unsigned int)(uintptr_t)&As[0][0][0];
  const unsigned int ldsB = (unsigned int)(uintptr_t)&Bs[0][0][0];

  f32x8 acc00 = {}, acc01 = {}, acc10 = {}, acc11 = {};

  auto issue_tile = [&](int k0, int buf) {
    #pragma unroll
    for (int i = 0; i < 4; ++i) {            // A: 128 rows, 4 x 32
      int row = crow + i * 32;
      unsigned int l = ldsA + (unsigned int)(((buf * 128 + row) * 64 + ccol) * 2);
      async_copy16(l, X + (size_t)(mBlock + row) * Kd + k0 + ccol);
    }
    #pragma unroll
    for (int i = 0; i < 2; ++i) {            // B: 64 rows, 2 x 32
      int row = crow + i * 32;
      int nrow = nBlock + row;
      int safe = nrow < N ? nrow : (N - 1);  // clamp; cols >= N never stored
      unsigned int l = ldsB + (unsigned int)(((buf * 64 + row) * 64 + ccol) * 2);
      async_copy16(l, W + (size_t)safe * Kd + k0 + ccol);
    }
  };

  // prologue: DMA tile 0
  issue_tile(0, 0);
  wait_async0();
  __syncthreads();

  const int nk = Kd / 64;
  for (int k = 0; k < nk; ++k) {
    const int cur = k & 1, nxt = cur ^ 1;
    if (k + 1 < nk) issue_tile((k + 1) * 64, nxt);   // overlap DMA with WMMA

    #pragma unroll
    for (int kk = 0; kk < 64; kk += 32) {
      FragU a0, a1, b0, b1;
      const int kb = kk + lane_hi * 16;
      a0.u = *(const uint8v*)&As[cur][mWave      + lane_lo][kb];
      a1.u = *(const uint8v*)&As[cur][mWave + 16 + lane_lo][kb];
      b0.u = *(const uint8v*)&Bs[cur][nWave      + lane_lo][kb];
      b1.u = *(const uint8v*)&Bs[cur][nWave + 16 + lane_lo][kb];
      acc00 = __builtin_amdgcn_wmma_f32_16x16x32_bf16(
          false, a0.v, false, b0.v, (short)0, acc00, false, false);
      acc01 = __builtin_amdgcn_wmma_f32_16x16x32_bf16(
          false, a0.v, false, b1.v, (short)0, acc01, false, false);
      acc10 = __builtin_amdgcn_wmma_f32_16x16x32_bf16(
          false, a1.v, false, b0.v, (short)0, acc10, false, false);
      acc11 = __builtin_amdgcn_wmma_f32_16x16x32_bf16(
          false, a1.v, false, b1.v, (short)0, acc11, false, false);
    }
    wait_async0();       // own wave's DMA for tile k+1 done
    __syncthreads();     // all waves' slices visible; tile k fully consumed
  }

  // ---- store D: VGPR r holds (m = r + 8*lane_hi, n = lane_lo) ----
  const int c0 = nBlock + nWave + lane_lo;
  const int c1 = c0 + 16;
  #pragma unroll
  for (int r = 0; r < 8; ++r) {
    int row0 = mBlock + mWave + r + 8 * lane_hi;
    int row1 = row0 + 16;
    if (c0 < N) {
      Y[(size_t)row0 * N + c0] = acc00[r];
      Y[(size_t)row1 * N + c0] = acc10[r];
    }
    if (c1 < N) {
      Y[(size_t)row0 * N + c1] = acc01[r];
      Y[(size_t)row1 * N + c1] = acc11[r];
    }
  }
}

// =====================================================================
// depthwise causal conv (K=4) + SiLU, layout (B,T,CONVC)
// =====================================================================
__global__ void conv_silu_kernel(const float* __restrict__ qkv,
                                 const float* __restrict__ conv_w,
                                 float* __restrict__ qkvc) {
  size_t idx = (size_t)blockIdx.x * blockDim.x + threadIdx.x;
  const size_t total = (size_t)BB * TT * CONVC;
  if (idx >= total) return;
  int c = (int)(idx % CONVC);
  int t = (int)((idx / CONVC) % TT);
  int b = (int)(idx / ((size_t)CONVC * TT));
  const float* w = conv_w + (size_t)c * KW;
  float acc = 0.f;
  #pragma unroll
  for (int j = 0; j < KW; ++j) {
    int tt = t + j - (KW - 1);
    if (tt >= 0) acc += w[j] * qkv[((size_t)(b * TT + tt)) * CONVC + c];
  }
  qkvc[idx] = acc / (1.f + __expf(-acc));   // silu
}

// new_conv_buf[b][c][j] = qkv[b][T-3+j][c],  shape (B, CONVC, KW-1)
__global__ void conv_tail_kernel(const float* __restrict__ qkv,
                                 float* __restrict__ buf) {
  int idx = blockIdx.x * blockDim.x + threadIdx.x;
  const int total = BB * CONVC * (KW - 1);
  if (idx >= total) return;
  int j = idx % (KW - 1);
  int c = (idx / (KW - 1)) % CONVC;
  int b = idx / ((KW - 1) * CONVC);
  buf[idx] = qkv[((size_t)(b * TT + (TT - (KW - 1) + j))) * CONVC + c];
}

// =====================================================================
// gated delta-rule scan: one workgroup per (b, v-head). 128x128 fp32
// state lives in LDS (64KB). 256 threads; thread = (j = tid&127,
// half = tid>>7) owns a half-column of the state each pass.
// =====================================================================
__global__ __launch_bounds__(256)
void scan_kernel(const float* __restrict__ qkvc,
                 const float* __restrict__ braw,
                 const float* __restrict__ araw,
                 const float* __restrict__ dt_bias,
                 const float* __restrict__ A_log,
                 float* __restrict__ o_buf,
                 float* __restrict__ state_out) {
  extern __shared__ float smem[];
  float* st   = smem;                 // [HD][HD]
  float* qs   = st + HD * HD;         // [HD]
  float* ks   = qs + HD;
  float* vs   = ks + HD;
  float* dl   = vs + HD;
  float* part = dl + HD;              // [2][HD]
  float* red  = part + 2 * HD;        // [8]
  float* sc   = red + 8;              // beta, decay

  const int tid  = threadIdx.x;
  const int bh   = blockIdx.x;
  const int b    = bh / NVH;
  const int h    = bh % NVH;
  const int kh   = h / EXP_;
  const int j    = tid & (HD - 1);
  const int half = tid >> 7;
  const int i0   = half * 64;

  for (int i = tid; i < HD * HD; i += 256) st[i] = 0.f;
  const float aexp = -__expf(A_log[h]);
  const float dtb  = dt_bias[h];
  __syncthreads();

  for (int t = 0; t < TT; ++t) {
    const float* base = qkvc + ((size_t)(b * TT + t)) * CONVC;
    if (tid < HD) {
      float qv = base[kh * HD + tid];
      float kv = base[KDIM + kh * HD + tid];
      vs[tid]  = base[2 * KDIM + h * HD + tid];
      qs[tid] = qv; ks[tid] = kv;
      float s1 = qv * qv, s2 = kv * kv;
      #pragma unroll
      for (int m = 16; m >= 1; m >>= 1) {
        s1 += __shfl_xor(s1, m, 32);
        s2 += __shfl_xor(s2, m, 32);
      }
      if ((tid & 31) == 0) { red[tid >> 5] = s1; red[4 + (tid >> 5)] = s2; }
    }
    if (tid == 0) {
      float br = braw[(size_t)(b * TT + t) * NVH + h];
      float ar = araw[(size_t)(b * TT + t) * NVH + h];
      sc[0] = 1.f / (1.f + __expf(-br));                // beta
      float xx = ar + dtb;
      float sp = (xx > 20.f) ? xx : log1pf(__expf(xx)); // softplus
      sc[1] = __expf(aexp * sp);                        // decay
    }
    __syncthreads();
    if (tid < HD) {
      float qq = red[0] + red[1] + red[2] + red[3];
      float kk = red[4] + red[5] + red[6] + red[7];
      qs[tid] = qs[tid] / fmaxf(sqrtf(qq), 1e-12f);
      ks[tid] = ks[tid] / fmaxf(sqrtf(kk), 1e-12f);
    }
    __syncthreads();

    const float beta = sc[0], decay = sc[1];
    // pass 1: state *= decay, partial Sk[j] = sum_i k[i]*state[i][j]
    float p = 0.f;
    #pragma unroll 4
    for (int i = i0; i < i0 + 64; ++i) {
      float s = st[i * HD + j] * decay;
      st[i * HD + j] = s;
      p += ks[i] * s;
    }
    part[half * HD + j] = p;
    __syncthreads();
    if (tid < HD) {
      float Sk = part[tid] + part[HD + tid];
      dl[tid] = (vs[tid] - Sk) * beta;                  // delta
    }
    __syncthreads();
    // pass 2: state += outer(k, delta), partial o[j] = sum_i q[i]*state
    p = 0.f;
    const float dj = dl[j];
    #pragma unroll 4
    for (int i = i0; i < i0 + 64; ++i) {
      float s = st[i * HD + j] + ks[i] * dj;
      st[i * HD + j] = s;
      p += qs[i] * s;
    }
    part[half * HD + j] = p;
    __syncthreads();
    if (tid < HD) {
      o_buf[(((size_t)(b * TT + t)) * NVH + h) * HD + tid] =
          part[tid] + part[HD + tid];
    }
    __syncthreads();
  }

  // final state -> d_out section (B, NVH, HD, HD)
  for (int idx = tid; idx < HD * HD; idx += 256) {
    state_out[((size_t)(b * NVH + h)) * HD * HD + idx] = st[idx];
  }
}

// =====================================================================
// y = o * silu(z); RMS-norm over HD with norm_w -> bf16 y for final GEMM.
// One 128-thread block per (b,t,h).
// =====================================================================
__global__ __launch_bounds__(128)
void ynorm_kernel(const float* __restrict__ o_buf, const float* __restrict__ z,
                  const float* __restrict__ norm_w, bf16* __restrict__ yb) {
  __shared__ float red[4];
  const int bh = blockIdx.x;            // (b*T+t)*NVH + h
  const int d  = threadIdx.x;
  const size_t base = (size_t)bh * HD;
  float zv = z[base + d];
  float y  = o_buf[base + d] * (zv / (1.f + __expf(-zv)));
  float s  = y * y;
  #pragma unroll
  for (int m = 16; m >= 1; m >>= 1) s += __shfl_xor(s, m, 32);
  if ((d & 31) == 0) red[d >> 5] = s;
  __syncthreads();
  float ms = (red[0] + red[1] + red[2] + red[3]) * (1.f / HD);
  yb[base + d] = (bf16)(y * rsqrtf(ms + 1e-6f) * norm_w[d]);
}

// =====================================================================
// launch
// =====================================================================
extern "C" void kernel_launch(void* const* d_in, const int* in_sizes, int n_in,
                              void* d_out, int out_size, void* d_ws,
                              size_t ws_size, hipStream_t stream) {
  (void)in_sizes; (void)n_in; (void)out_size; (void)ws_size;
  const float* x      = (const float*)d_in[0];
  const float* W_qkv  = (const float*)d_in[1];
  const float* W_z    = (const float*)d_in[2];
  const float* W_b    = (const float*)d_in[3];
  const float* W_a    = (const float*)d_in[4];
  const float* conv_w = (const float*)d_in[5];
  const float* dt_b   = (const float*)d_in[6];
  const float* A_log  = (const float*)d_in[7];
  const float* norm_w = (const float*)d_in[8];
  const float* W_out  = (const float*)d_in[9];

  float* out = (float*)d_out;                      // (B,T,HID)
  float* state_out = out + (size_t)BB * TT * HID;  // (B,NVH,HD,HD)
  float* convbuf   = state_out + (size_t)BB * NVH * HD * HD; // (B,CONVC,3)

  // ---- workspace carve-up ----
  float* ws    = (float*)d_ws;
  float* qkv   = ws;                                  // fp32, 33.5M elems
  float* qkvc  = qkv  + (size_t)MROWS * CONVC;        // fp32, 33.5M
  float* z     = qkvc + (size_t)MROWS * CONVC;        // fp32, 16.8M
  float* brw   = z    + (size_t)MROWS * VDIM;         // fp32, 128K
  float* arw   = brw  + (size_t)MROWS * NVH;          // fp32, 128K
  float* obuf  = arw  + (size_t)MROWS * NVH;          // fp32, 16.8M
  bf16* xb     = (bf16*)(obuf + (size_t)MROWS * VDIM);
  bf16* Wqkvb  = xb    + (size_t)MROWS * HID;
  bf16* Wzb    = Wqkvb + (size_t)CONVC * HID;
  bf16* Wbb    = Wzb   + (size_t)VDIM * HID;
  bf16* Wab    = Wbb   + (size_t)NVH * HID;
  bf16* Woutb  = Wab   + (size_t)NVH * HID;
  bf16* yb     = Woutb + (size_t)HID * VDIM;

  const dim3 blk256(256);
  auto cvt = [&](const float* src, bf16* dst, size_t n) {
    cvt_bf16_kernel<<<dim3((unsigned)((n / 4 + 255) / 256)), blk256, 0,
                      stream>>>(src, dst, n);
  };

  // one-time fp32 -> bf16 packs (weights then stay L2-resident as bf16)
  cvt(x,     xb,    (size_t)MROWS * HID);
  cvt(W_qkv, Wqkvb, (size_t)CONVC * HID);
  cvt(W_z,   Wzb,   (size_t)VDIM * HID);
  cvt(W_b,   Wbb,   (size_t)NVH * HID);
  cvt(W_a,   Wab,   (size_t)NVH * HID);
  cvt(W_out, Woutb, (size_t)HID * VDIM);

  // projections (async-DMA double-buffered bf16 WMMA GEMMs)
  gemm_bf16_wmma<<<dim3(CONVC / 64, MROWS / 128), blk256, 0, stream>>>(
      xb, Wqkvb, qkv, MROWS, CONVC, HID);
  gemm_bf16_wmma<<<dim3(VDIM / 64, MROWS / 128), blk256, 0, stream>>>(
      xb, Wzb, z, MROWS, VDIM, HID);
  gemm_bf16_wmma<<<dim3(1, MROWS / 128), blk256, 0, stream>>>(
      xb, Wbb, brw, MROWS, NVH, HID);
  gemm_bf16_wmma<<<dim3(1, MROWS / 128), blk256, 0, stream>>>(
      xb, Wab, arw, MROWS, NVH, HID);

  // conv + silu, and new_conv_buf tail
  {
    size_t total = (size_t)BB * TT * CONVC;
    conv_silu_kernel<<<dim3((unsigned)((total + 255) / 256)), blk256, 0,
                       stream>>>(qkv, conv_w, qkvc);
    int tail = BB * CONVC * (KW - 1);
    conv_tail_kernel<<<dim3((tail + 255) / 256), blk256, 0, stream>>>(
        qkv, convbuf);
  }

  // sequential gated delta scan (64 head-workgroups, 68.6KB dynamic LDS)
  {
    size_t lds = (size_t)(HD * HD + 4 * HD + 2 * HD + 8 + 2) * sizeof(float);
    scan_kernel<<<dim3(BB * NVH), blk256, lds, stream>>>(
        qkvc, brw, arw, dt_b, A_log, obuf, state_out);
  }

  // gated RMS norm -> bf16 y
  ynorm_kernel<<<dim3(BB * TT * NVH), dim3(128), 0, stream>>>(
      obuf, z, norm_w, yb);

  // output projection
  gemm_bf16_wmma<<<dim3(HID / 64, MROWS / 128), blk256, 0, stream>>>(
      yb, Woutb, out, MROWS, HID, VDIM);
}
